// GPT_87093346828299
// MI455X (gfx1250) — compile-verified, matrix operand order checked by reading, not run on hardware
//
#include <hip/hip_runtime.h>
#include <cstdint>
#include <cstddef>

// ---- model dims (fixed by the reference) ----
#define Bsz  2
#define Tlen 4096
#define Cd   512
#define Hh   8
#define Nd   64
#define Ll   6
#define Qc   256
#define NC   (Tlen / Qc)      // 16 chunks
#define Vv   32000
#define C3   1536
#define LN_EPS 1e-5f
#define GN_EPS (1e-5f * 64.0f)

typedef __attribute__((ext_vector_type(16))) __bf16 v16bf;
typedef __attribute__((ext_vector_type(8)))  float  v8f;

union FragBF { v16bf v; unsigned u[8]; };

__device__ __forceinline__ unsigned short f2bf(float f) {
  unsigned x = __builtin_bit_cast(unsigned, f);
  x = x + 0x7FFFu + ((x >> 16) & 1u);   // round-to-nearest-even
  return (unsigned short)(x >> 16);
}

// ============================================================
// GEMM: Y[M,N] = A[M,K] @ W[N,K]^T   (bf16 in, f32 out)
// Block 256 threads (8 waves), tile 128(M) x 64(N); wave tile 32x32.
// Double-buffered LDS staged via GLOBAL_LOAD_ASYNC_TO_LDS_B128:
// tile k+1 DMA overlaps WMMA on tile k.
// ============================================================
__global__ __launch_bounds__(256) void k_gemm(const unsigned short* __restrict__ A,
                                              const unsigned short* __restrict__ W,
                                              float* __restrict__ Y,
                                              int M, int N, int K) {
  __shared__ unsigned Asu[2][128 * 20];   // 2 x 10,240 B (16B-aligned slots)
  __shared__ unsigned Bsu[2][64 * 20];    // 2 x  5,120 B

  const int tid   = threadIdx.x;
  const int lane  = tid & 31;
  const int wid   = tid >> 5;
  const int waveM = wid & 3;
  const int waveN = wid >> 2;
  const int lrow  = lane & 15;
  const int lhi   = lane >> 4;
  const int bM    = blockIdx.y * 128;
  const int bN    = blockIdx.x * 64;

  // staging: A = 128 rows x 4 x 16B (2 slots/thread), B = 64 x 4 x 16B (1 slot/thread)
  const int ra  = tid >> 1;
  const int qa0 = (tid & 1) * 2;
  const int qa1 = qa0 + 1;
  const int rb  = tid >> 2;
  const int qb  = tid & 3;

  const unsigned lA0 = (unsigned)(uintptr_t)&Asu[0][ra * 20 + qa0 * 4];
  const unsigned lA1 = (unsigned)(uintptr_t)&Asu[0][ra * 20 + qa1 * 4];
  const unsigned lB  = (unsigned)(uintptr_t)&Bsu[0][rb * 20 + qb * 4];
  const bool mok = (bM + ra < M);
  const bool nok = (bN + rb < N);
  const size_t baseA = (size_t)(bM + ra) * K;
  const size_t baseB = (size_t)(bN + rb) * K;

  auto issue = [&](int k0, int buf) {
    unsigned offA = (unsigned)buf * (128u * 20u * 4u);
    unsigned offB = (unsigned)buf * (64u * 20u * 4u);
    if (mok) {
      unsigned g0 = (unsigned)((baseA + k0 + qa0 * 8) * 2);
      unsigned l0 = lA0 + offA;
      asm volatile("global_load_async_to_lds_b128 %0, %1, %2"
                   :: "v"(l0), "v"(g0), "s"(A) : "memory");
      unsigned g1 = (unsigned)((baseA + k0 + qa1 * 8) * 2);
      unsigned l1 = lA1 + offA;
      asm volatile("global_load_async_to_lds_b128 %0, %1, %2"
                   :: "v"(l1), "v"(g1), "s"(A) : "memory");
    }
    if (nok) {
      unsigned gb = (unsigned)((baseB + k0 + qb * 8) * 2);
      unsigned lb = lB + offB;
      asm volatile("global_load_async_to_lds_b128 %0, %1, %2"
                   :: "v"(lb), "v"(gb), "s"(W) : "memory");
    }
  };

  v8f acc00 = {}, acc01 = {}, acc10 = {}, acc11 = {};

  issue(0, 0);
  asm volatile("s_wait_asynccnt 0x0" ::: "memory");
  __syncthreads();

  const int nk = K >> 5;
  for (int kk = 0; kk < nk; ++kk) {
    const int cur = kk & 1;
    if (kk + 1 < nk) issue((kk + 1) * 32, cur ^ 1);   // overlap DMA with WMMA

    const unsigned* As = Asu[cur];
    const unsigned* Bs = Bsu[cur];
    FragBF fa0, fa1, fb0, fb1;
#pragma unroll
    for (int vI = 0; vI < 8; ++vI) {
      int kp = (vI >> 2) * 8 + lhi * 4 + (vI & 3);
      fa0.u[vI] = As[(waveM * 32      + lrow) * 20 + kp];
      fa1.u[vI] = As[(waveM * 32 + 16 + lrow) * 20 + kp];
      fb0.u[vI] = Bs[(waveN * 32      + lrow) * 20 + kp];
      fb1.u[vI] = Bs[(waveN * 32 + 16 + lrow) * 20 + kp];
    }
    acc00 = __builtin_amdgcn_wmma_f32_16x16x32_bf16(false, fa0.v, false, fb0.v, (short)0, acc00, false, false);
    acc01 = __builtin_amdgcn_wmma_f32_16x16x32_bf16(false, fa0.v, false, fb1.v, (short)0, acc01, false, false);
    acc10 = __builtin_amdgcn_wmma_f32_16x16x32_bf16(false, fa1.v, false, fb0.v, (short)0, acc10, false, false);
    acc11 = __builtin_amdgcn_wmma_f32_16x16x32_bf16(false, fa1.v, false, fb1.v, (short)0, acc11, false, false);

    asm volatile("s_wait_asynccnt 0x0" ::: "memory");
    __syncthreads();
  }

#pragma unroll
  for (int vI = 0; vI < 8; ++vI) {
    int gm0 = bM + waveM * 32 + vI + 8 * lhi;
    int gm1 = gm0 + 16;
    int gn0 = bN + waveN * 32 + lrow;
    int gn1 = gn0 + 16;
    if (gm0 < M) {
      if (gn0 < N) Y[(size_t)gm0 * N + gn0] = acc00[vI];
      if (gn1 < N) Y[(size_t)gm0 * N + gn1] = acc01[vI];
    }
    if (gm1 < M) {
      if (gn0 < N) Y[(size_t)gm1 * N + gn0] = acc10[vI];
      if (gn1 < N) Y[(size_t)gm1 * N + gn1] = acc11[vI];
    }
  }
}

// ============================================================
// Embedding
// ============================================================
__global__ __launch_bounds__(256) void k_embed(const int* __restrict__ idx,
                                               const float* __restrict__ wte,
                                               const float* __restrict__ wpe,
                                               float* __restrict__ x) {
  int row = blockIdx.x;
  int t   = row % Tlen;
  int tok = idx[row];
  int e0  = threadIdx.x;
  const float* wt = wte + (size_t)tok * Cd;
  const float* wp = wpe + (size_t)t * Cd;
  float* xo = x + (size_t)row * Cd;
  xo[e0]       = wt[e0]       + wp[e0];
  xo[e0 + 256] = wt[e0 + 256] + wp[e0 + 256];
}

// ============================================================
// LayerNorm (no bias)
// ============================================================
__global__ __launch_bounds__(256) void k_layernorm(const float* __restrict__ x,
                                                   const float* __restrict__ w,
                                                   float* __restrict__ out,
                                                   long long rstride, float eps) {
  __shared__ float red[256];
  int row = blockIdx.x;
  int tid = threadIdx.x;
  const float* xr = x + (size_t)row * (size_t)rstride;
  float v0 = xr[tid], v1 = xr[tid + 256];

  red[tid] = v0 + v1;
  __syncthreads();
  for (int s = 128; s > 0; s >>= 1) {
    if (tid < s) red[tid] += red[tid + s];
    __syncthreads();
  }
  float mu = red[0] * (1.f / 512.f);
  __syncthreads();
  float d0 = v0 - mu, d1 = v1 - mu;
  red[tid] = d0 * d0 + d1 * d1;
  __syncthreads();
  for (int s = 128; s > 0; s >>= 1) {
    if (tid < s) red[tid] += red[tid + s];
    __syncthreads();
  }
  float rs = rsqrtf(red[0] * (1.f / 512.f) + eps);
  float* o = out + (size_t)row * Cd;
  o[tid]       = d0 * rs * w[tid];
  o[tid + 256] = d1 * rs * w[tid + 256];
}

// ============================================================
// Token-shift mix -> bf16
// ============================================================
__global__ __launch_bounds__(256) void k_mix_bf(const float* __restrict__ xln,
                                                const float* __restrict__ maa,
                                                unsigned short* __restrict__ out) {
  int row = blockIdx.x;
  int t   = row % Tlen;
  const float* cur = xln + (size_t)row * Cd;
  const float* prv = cur - Cd;
  unsigned short* o = out + (size_t)row * Cd;
  int e = threadIdx.x;
#pragma unroll
  for (int h = 0; h < 2; ++h, e += 256) {
    float xc = cur[e];
    float xp = (t == 0) ? 0.f : prv[e];
    o[e] = f2bf(xc + (xp - xc) * maa[e]);
  }
}

// ============================================================
// WKV pass A: per-chunk state contribution (parallel over b,h,chunk)
//   G_c[m,n] = sum_j k[j,m] * wdec^(Q-1-j) * v[j,n]
// ============================================================
__global__ __launch_bounds__(256) void k_wkv_state(const float* __restrict__ k,
                                                   const float* __restrict__ v,
                                                   const float* __restrict__ tdecay,
                                                   float* __restrict__ G) {
  __shared__ float kT[64][64];
  __shared__ float vT[64][64];
  __shared__ float pw[Qc];

  const int blk = blockIdx.x;          // bh*NC + c
  const int c   = blk % NC;
  const int bh  = blk / NC;
  const int b   = bh / Hh, h = bh % Hh;
  const int tid = threadIdx.x;
  const float wdec = expf(-expf(tdecay[h]));

  for (int i2 = tid; i2 < Qc; i2 += 256) pw[i2] = powf(wdec, (float)i2);

  const int m_own = tid >> 2;
  const int n_own = (tid & 3) * 16;
  float s2[16];
#pragma unroll
  for (int q = 0; q < 16; ++q) s2[q] = 0.f;

  const int t0 = c * Qc;
#pragma unroll 1
  for (int jt = 0; jt < 4; ++jt) {
    __syncthreads();
    {
      int jj = tid >> 2;
      int g  = (tid & 3) * 16;
      size_t base = ((size_t)b * Tlen + t0 + jt * 64 + jj) * Cd + h * Nd + g;
#pragma unroll
      for (int q2 = 0; q2 < 16; ++q2) { kT[jj][g + q2] = k[base + q2]; vT[jj][g + q2] = v[base + q2]; }
    }
    __syncthreads();
#pragma unroll 1
    for (int jj = 0; jj < 64; ++jj) {
      int j = jt * 64 + jj;
      float km = kT[jj][m_own] * pw[Qc - 1 - j];
#pragma unroll
      for (int q = 0; q < 16; ++q) s2[q] += km * vT[jj][n_own + q];
    }
  }
  float* Gp = G + ((size_t)bh * NC + c) * 4096;
#pragma unroll
  for (int q = 0; q < 16; ++q) Gp[m_own * 64 + n_own + q] = s2[q];
}

// ============================================================
// WKV pass B: sequential scan of chunk states (tiny)
//   S_0 = 0 ; S_c = ws*S_{c-1} + G_{c-1}
// ============================================================
__global__ __launch_bounds__(256) void k_wkv_scan(const float* __restrict__ G,
                                                  const float* __restrict__ tdecay,
                                                  float* __restrict__ S) {
  const int bh = blockIdx.x;
  const int h  = bh % Hh;
  const int tid = threadIdx.x;
  const float wdec = expf(-expf(tdecay[h]));
  const float ws   = powf(wdec, (float)Qc);
  float s[16];
#pragma unroll
  for (int q = 0; q < 16; ++q) s[q] = 0.f;
  for (int c = 0; c < NC; ++c) {
    float*       Sp = S + ((size_t)bh * NC + c) * 4096 + tid * 16;
    const float* Gp = G + ((size_t)bh * NC + c) * 4096 + tid * 16;
#pragma unroll
    for (int q = 0; q < 16; ++q) { Sp[q] = s[q]; s[q] = ws * s[q] + Gp[q]; }
  }
}

// ============================================================
// WKV pass C: outputs (parallel over b,h,chunk); one thread per row i
//   y[i,:] = sum_{j<=i} W[i,j] (r_i . k_j) v_j  +  wdec^i * (r_i @ S_c)
// ============================================================
__global__ __launch_bounds__(256) void k_wkv_y(const float* __restrict__ r,
                                               const float* __restrict__ k,
                                               const float* __restrict__ v,
                                               const float* __restrict__ S,
                                               const float* __restrict__ tdecay,
                                               const float* __restrict__ tfaaaa,
                                               float* __restrict__ y) {
  __shared__ float Ssh[64][64];
  __shared__ float kT[64][64];
  __shared__ float vT[64][64];
  __shared__ float pw[Qc];

  const int blk = blockIdx.x;
  const int c   = blk % NC;
  const int bh  = blk / NC;
  const int b   = bh / Hh, h = bh % Hh;
  const int tid = threadIdx.x;
  const float wdec = expf(-expf(tdecay[h]));
  const float u    = tfaaaa[h];

  for (int i2 = tid; i2 < Qc; i2 += 256) pw[i2] = powf(wdec, (float)i2);
  {
    const float* Sp = S + ((size_t)bh * NC + c) * 4096;
    for (int i2 = tid; i2 < 4096; i2 += 256) ((float*)Ssh)[i2] = Sp[i2];
  }
  __syncthreads();

  const int i  = tid;
  const int t0 = c * Qc;
  const float pwi = pw[i];

  float ri[64];
  const float* rp = r + ((size_t)b * Tlen + t0 + i) * Cd + h * Nd;
#pragma unroll
  for (int m = 0; m < 64; ++m) ri[m] = rp[m];

  float yacc[64];
#pragma unroll
  for (int n = 0; n < 64; ++n) yacc[n] = 0.f;

#pragma unroll 1
  for (int jt = 0; jt < 4; ++jt) {
    __syncthreads();
    {
      int jj = tid >> 2;
      int g  = (tid & 3) * 16;
      size_t base = ((size_t)b * Tlen + t0 + jt * 64 + jj) * Cd + h * Nd + g;
#pragma unroll
      for (int q2 = 0; q2 < 16; ++q2) { kT[jj][g + q2] = k[base + q2]; vT[jj][g + q2] = v[base + q2]; }
    }
    __syncthreads();
#pragma unroll 1
    for (int jj = 0; jj < 64; ++jj) {
      int j = jt * 64 + jj;
      float wij = (j < i) ? pw[i - j - 1] : ((j == i) ? u : 0.f);
      if (wij != 0.f) {
        float a = 0.f;
#pragma unroll
        for (int m = 0; m < 64; ++m) a += ri[m] * kT[jj][m];
        a *= wij;
#pragma unroll
        for (int n = 0; n < 64; ++n) yacc[n] += a * vT[jj][n];
      }
    }
  }

#pragma unroll
  for (int m = 0; m < 64; ++m) {
    float cm = ri[m] * pwi;
#pragma unroll
    for (int n = 0; n < 64; ++n) yacc[n] += cm * Ssh[m][n];
  }

  float* yp = y + ((size_t)b * Tlen + t0 + i) * Cd + h * Nd;
#pragma unroll
  for (int n = 0; n < 64; ++n) yp[n] = yacc[n];
}

// ============================================================
// GroupNorm(H groups) + gate -> bf16
// ============================================================
__global__ __launch_bounds__(256) void k_gn_gate(const float* __restrict__ y,
                                                 const float* __restrict__ g,
                                                 const float* __restrict__ gnw,
                                                 const float* __restrict__ gnb,
                                                 unsigned short* __restrict__ out) {
  __shared__ float buf[512];
  __shared__ float hmu[8], hsq[8];
  int row = blockIdx.x;
  int tid = threadIdx.x;
  const float* yr = y + (size_t)row * Cd;
  float v0 = yr[tid], v1 = yr[tid + 256];
  buf[tid] = v0; buf[tid + 256] = v1;
  __syncthreads();
  if (tid < 8) {
    float s = 0.f, q = 0.f;
#pragma unroll
    for (int n = 0; n < 64; ++n) { float z = buf[tid * 64 + n]; s += z; q += z * z; }
    hmu[tid] = s * (1.f / 64.f);
    hsq[tid] = q * (1.f / 64.f);
  }
  __syncthreads();
  const float* gr = g + (size_t)row * Cd;
  unsigned short* o = out + (size_t)row * Cd;
  int e = tid;
  float vv = v0;
#pragma unroll
  for (int hrep = 0; hrep < 2; ++hrep, e += 256) {
    int hd = e >> 6;
    float mu  = hmu[hd];
    float var = hsq[hd] - mu * mu;
    float rs  = rsqrtf(var + GN_EPS);
    float yn  = (vv - mu) * rs * gnw[e] + gnb[e];
    float gg  = gr[e];
    o[e] = f2bf(yn * (gg / (1.f + expf(-gg))));
    vv = v1;
  }
}

// ============================================================
// elementwise helpers
// ============================================================
__global__ __launch_bounds__(256) void k_add(float* __restrict__ x,
                                             const float* __restrict__ a, size_t n) {
  size_t i = (size_t)blockIdx.x * blockDim.x + threadIdx.x;
  size_t st = (size_t)gridDim.x * blockDim.x;
  for (; i < n; i += st) x[i] += a[i];
}
__global__ __launch_bounds__(256) void k_relusq_bf(const float* __restrict__ in,
                                                   unsigned short* __restrict__ out, size_t n) {
  size_t i = (size_t)blockIdx.x * blockDim.x + threadIdx.x;
  size_t st = (size_t)gridDim.x * blockDim.x;
  for (; i < n; i += st) { float v = in[i]; v = v > 0.f ? v : 0.f; out[i] = f2bf(v * v); }
}
__global__ __launch_bounds__(256) void k_sigmul_add(float* __restrict__ x,
                                                    const float* __restrict__ s,
                                                    const float* __restrict__ h2, size_t n) {
  size_t i = (size_t)blockIdx.x * blockDim.x + threadIdx.x;
  size_t st = (size_t)gridDim.x * blockDim.x;
  for (; i < n; i += st) x[i] += h2[i] / (1.f + expf(-s[i]));
}
__global__ __launch_bounds__(256) void k_f2bf(const float* __restrict__ in,
                                              unsigned short* __restrict__ out, size_t n) {
  size_t i = (size_t)blockIdx.x * blockDim.x + threadIdx.x;
  size_t st = (size_t)gridDim.x * blockDim.x;
  for (; i < n; i += st) out[i] = f2bf(in[i]);
}

// ============================================================
// host-side launcher
// ============================================================
static inline void gemm(const unsigned short* A, const unsigned short* W, float* Y,
                        int M, int N, int K, hipStream_t s) {
  dim3 grid((N + 63) / 64, (M + 127) / 128);
  k_gemm<<<grid, 256, 0, s>>>(A, W, Y, M, N, K);
}

extern "C" void kernel_launch(void* const* d_in, const int* in_sizes, int n_in,
                              void* d_out, int out_size, void* d_ws, size_t ws_size,
                              hipStream_t stream) {
  (void)in_sizes; (void)n_in; (void)out_size; (void)ws_size;

  const int*   idx    = (const int*)  d_in[0];
  const float* wte    = (const float*)d_in[1];
  const float* wpe    = (const float*)d_in[2];
  const float* ln1_w  = (const float*)d_in[3];
  const float* ln2_w  = (const float*)d_in[4];
  const float* lnf_w  = (const float*)d_in[5];
  const float* maa_k  = (const float*)d_in[6];
  const float* maa_v  = (const float*)d_in[7];
  const float* maa_r  = (const float*)d_in[8];
  const float* maa_g  = (const float*)d_in[9];
  const float* tdecay = (const float*)d_in[10];
  const float* tfaaaa = (const float*)d_in[11];
  const float* Wr     = (const float*)d_in[12];
  const float* Wk     = (const float*)d_in[13];
  const float* Wv     = (const float*)d_in[14];
  const float* Wg     = (const float*)d_in[15];
  const float* Wo     = (const float*)d_in[16];
  const float* gn_w   = (const float*)d_in[17];
  const float* gn_b   = (const float*)d_in[18];
  const float* cmaa_k = (const float*)d_in[19];
  const float* cmaa_r = (const float*)d_in[20];
  const float* Wck    = (const float*)d_in[21];
  const float* Wcv    = (const float*)d_in[22];
  const float* Wcr    = (const float*)d_in[23];

  const size_t SZ   = (size_t)Bsz * Tlen * Cd;    // 4,194,304 floats
  const size_t SZ32 = SZ / 2 * 3;
  float* ws = (float*)d_ws;

  float*          x     = ws;
  float*          xln   = ws + SZ;
  unsigned short* mixbf = (unsigned short*)(ws + 2 * SZ);   // bf16 A operands (up to B*T*3C)
  float*          big   = ws + 2 * SZ + SZ32;               // 4*SZ: r,k,v,g / h1 / wte-bf16
  float*          ybuf  = big + 4 * SZ;
  float*          abuf  = ybuf + SZ;
  unsigned short* wbf   = (unsigned short*)(abuf + SZ);     // per-GEMM weight bf16 (<= 3C*C)
  float*          Gbuf  = abuf + SZ + SZ / 8;               // B*H*NC*4096 = 1M floats
  float*          Sbuf  = Gbuf + (size_t)Bsz * Hh * NC * 4096;

  float* rbp = big;
  float* kbp = big + 1 * SZ;
  float* vbp = big + 2 * SZ;
  float* gbp = big + 3 * SZ;

  const int ROWS = Bsz * Tlen;
  const size_t CC = (size_t)Cd * Cd;

  k_embed<<<ROWS, 256, 0, stream>>>(idx, wte, wpe, x);

  for (int l = 0; l < Ll; ++l) {
    const float* Wr_l  = Wr  + (size_t)l * CC;
    const float* Wk_l  = Wk  + (size_t)l * CC;
    const float* Wv_l  = Wv  + (size_t)l * CC;
    const float* Wg_l  = Wg  + (size_t)l * CC;
    const float* Wo_l  = Wo  + (size_t)l * CC;
    const float* Wck_l = Wck + (size_t)l * C3 * Cd;
    const float* Wcv_l = Wcv + (size_t)l * Cd * C3;
    const float* Wcr_l = Wcr + (size_t)l * CC;
    const float* td_l  = tdecay + l * Hh;
    const float* tf_l  = tfaaaa + l * Hh;

    // ---- time mix ----
    k_layernorm<<<ROWS, 256, 0, stream>>>(x, ln1_w + l * Cd, xln, (long long)Cd, LN_EPS);

    k_mix_bf<<<ROWS, 256, 0, stream>>>(xln, maa_r + l * Cd, mixbf);
    k_f2bf<<<512, 256, 0, stream>>>(Wr_l, wbf, CC);
    gemm(mixbf, wbf, rbp, ROWS, Cd, Cd, stream);

    k_mix_bf<<<ROWS, 256, 0, stream>>>(xln, maa_k + l * Cd, mixbf);
    k_f2bf<<<512, 256, 0, stream>>>(Wk_l, wbf, CC);
    gemm(mixbf, wbf, kbp, ROWS, Cd, Cd, stream);

    k_mix_bf<<<ROWS, 256, 0, stream>>>(xln, maa_v + l * Cd, mixbf);
    k_f2bf<<<512, 256, 0, stream>>>(Wv_l, wbf, CC);
    gemm(mixbf, wbf, vbp, ROWS, Cd, Cd, stream);

    k_mix_bf<<<ROWS, 256, 0, stream>>>(xln, maa_g + l * Cd, mixbf);
    k_f2bf<<<512, 256, 0, stream>>>(Wg_l, wbf, CC);
    gemm(mixbf, wbf, gbp, ROWS, Cd, Cd, stream);

    // ---- WKV: 3-pass chunk-parallel recurrence ----
    k_wkv_state<<<Bsz * Hh * NC, 256, 0, stream>>>(kbp, vbp, td_l, Gbuf);
    k_wkv_scan <<<Bsz * Hh,      256, 0, stream>>>(Gbuf, td_l, Sbuf);
    k_wkv_y    <<<Bsz * Hh * NC, 256, 0, stream>>>(rbp, kbp, vbp, Sbuf, td_l, tf_l, ybuf);

    k_gn_gate<<<ROWS, 256, 0, stream>>>(ybuf, gbp, gn_w + l * Cd, gn_b + l * Cd, mixbf);
    k_f2bf<<<512, 256, 0, stream>>>(Wo_l, wbf, CC);
    gemm(mixbf, wbf, abuf, ROWS, Cd, Cd, stream);
    k_add<<<2048, 256, 0, stream>>>(x, abuf, SZ);

    // ---- channel mix ----
    k_layernorm<<<ROWS, 256, 0, stream>>>(x, ln2_w + l * Cd, xln, (long long)Cd, LN_EPS);

    k_mix_bf<<<ROWS, 256, 0, stream>>>(xln, cmaa_k + l * Cd, mixbf);
    k_f2bf<<<1024, 256, 0, stream>>>(Wck_l, wbf, (size_t)C3 * Cd);
    gemm(mixbf, wbf, big, ROWS, C3, Cd, stream);            // h1 f32 in big
    k_relusq_bf<<<4096, 256, 0, stream>>>(big, mixbf, SZ * 3);
    k_f2bf<<<1024, 256, 0, stream>>>(Wcv_l, wbf, (size_t)Cd * C3);
    gemm(mixbf, wbf, ybuf, ROWS, Cd, C3, stream);           // h2

    k_mix_bf<<<ROWS, 256, 0, stream>>>(xln, cmaa_r + l * Cd, mixbf);
    k_f2bf<<<512, 256, 0, stream>>>(Wcr_l, wbf, CC);
    gemm(mixbf, wbf, abuf, ROWS, Cd, Cd, stream);           // s
    k_sigmul_add<<<2048, 256, 0, stream>>>(x, abuf, ybuf, SZ);
  }

  // ---- final LN (last token only) + logits ----
  k_layernorm<<<Bsz, 256, 0, stream>>>(x + (size_t)(Tlen - 1) * Cd, lnf_w, abuf,
                                       (long long)Tlen * Cd, LN_EPS);
  k_f2bf<<<4, 256, 0, stream>>>(abuf, mixbf, (size_t)Bsz * Cd);
  k_f2bf<<<2048, 256, 0, stream>>>(wte, (unsigned short*)big, (size_t)Vv * Cd);
  gemm(mixbf, (unsigned short*)big, (float*)d_out, Bsz, Vv, Cd, stream);
}
